// MultiHeadAttention_35270271435394
// MI455X (gfx1250) — matrix operand order repaired and checked by reading, not compiled, not run on hardware
//
#include <hip/hip_runtime.h>

// ---------------- problem constants ----------------
#define DM 1024      // d_model
#define NH 16        // heads
#define HD 64        // head dim
#define BB 2         // batch
#define SS 2048      // seq
#define MROWS (BB * SS)  // 4096 rows of x

typedef __bf16 bf16_t;
typedef bf16_t v16bf __attribute__((ext_vector_type(16)));
typedef bf16_t v8bf  __attribute__((ext_vector_type(8)));
typedef float  v8f   __attribute__((ext_vector_type(8)));

// -------- CDNA5 async global->LDS path (guarded; sync fallback) --------
#if defined(__has_builtin)
#if __has_builtin(__builtin_amdgcn_global_load_async_to_lds_b128) && \
    __has_builtin(__builtin_amdgcn_s_wait_asynccnt)
#define HAVE_ASYNC_LDS 1
#endif
#endif

// Builtin prototype (from hipcc diagnostic): pointee is GCC-vector int x4;
// global side is address-space 1, LDS side is address-space 3 (32-bit ptr).
typedef int v4i_t __attribute__((vector_size(16)));
typedef __attribute__((address_space(1))) v4i_t glob_v4i;
typedef __attribute__((address_space(3))) v4i_t lds_v4i;

static __device__ __forceinline__ void copy16_g2lds(const bf16_t* g, bf16_t* lds) {
#if defined(HAVE_ASYNC_LDS)
    // Per-lane 16B DMA into LDS, tracked by ASYNCcnt (emits global_load_async_to_lds_b128).
    // Generic LDS pointers carry the LDS byte offset in the low 32 bits.
    __builtin_amdgcn_global_load_async_to_lds_b128(
        (glob_v4i*)(uintptr_t)g,
        (lds_v4i*)(uint32_t)(uintptr_t)lds, 0, 0);
#else
    *(v8bf*)lds = *(const v8bf*)g;  // synchronous fallback: global load + ds store
#endif
}

static __device__ __forceinline__ void wait_async_le2() {
#if defined(HAVE_ASYNC_LDS)
    __builtin_amdgcn_s_wait_asynccnt(2);
#endif
}
static __device__ __forceinline__ void wait_async_zero() {
#if defined(HAVE_ASYNC_LDS)
    __builtin_amdgcn_s_wait_asynccnt(0);
#endif
}

// ---------------- WMMA fragment helpers ----------------
static __device__ __forceinline__ v16bf cat8(v8bf lo, v8bf hi) {
    return __builtin_shufflevector(lo, hi, 0, 1, 2, 3, 4, 5, 6, 7,
                                   8, 9, 10, 11, 12, 13, 14, 15);
}

// A fragment: 16x32 bf16 (MxK). Row-major source, leading dim lda (elements).
// Lane l (0..15) holds row l; h selects K sub-half per ISA layout:
//   elems 0..7  = K[h*8 .. h*8+7],  elems 8..15 = K[16+h*8 .. 16+h*8+7]
static __device__ __forceinline__ v16bf load_a(const bf16_t* a, int lda, int l, int h) {
    const bf16_t* p = a + (size_t)l * lda + h * 8;
    v8bf lo = *(const v8bf*)p;
    v8bf hi = *(const v8bf*)(p + 16);
    return cat8(lo, hi);
}

// B fragment: 32x16 (KxN) sourced from row-major Bt[N][K] (NT form, k-contig).
// Lane l = column N; elems 0..7 = K[h*8 .. h*8+7], elems 8..15 = K[16+h*8 .. 16+h*8+7] of row (n0+l) of Bt.
static __device__ __forceinline__ v16bf load_b_nt(const bf16_t* bt, int ldb, int l, int h) {
    const bf16_t* p = bt + (size_t)l * ldb + h * 8;
    v8bf lo = *(const v8bf*)p;
    v8bf hi = *(const v8bf*)(p + 16);
    return cat8(lo, hi);
}

static __device__ __forceinline__ v8f wmma_bf16(v16bf a, v16bf b, v8f c) {
    return __builtin_amdgcn_wmma_f32_16x16x32_bf16(false, a, false, b, (short)0, c,
                                                   false, false);
}

// ---------------- fp32 -> bf16 convert ----------------
__global__ void f2bf_kernel(const float* __restrict__ in, bf16_t* __restrict__ out, int n) {
    int i = blockIdx.x * blockDim.x + threadIdx.x;
    int stride = gridDim.x * blockDim.x;
    for (; i < n; i += stride) out[i] = (bf16_t)in[i];
}

// ---------------- NT GEMM: C[M,N] = A[M,K] * Bt[N,K]^T + bias ----------------
// 256 threads = 8 waves; block tile 128(M) x 128(N); each wave: 32x64 sub-tile.
// B tile (128 x 32) staged in LDS via async DMA, double-buffered.
// MODE 0: out = float row-major [M,N]
// MODE 1: out = bf16 head-split [b, h, s, d]   (for Q, K)
// MODE 2: out = bf16 head-split transposed [b, h, d, s]  (for V^T)
template <int MODE>
__global__ __launch_bounds__(256) void gemm_nt_kernel(const bf16_t* __restrict__ A,
                                                      const bf16_t* __restrict__ Bt,
                                                      const float* __restrict__ bias,
                                                      void* __restrict__ outp,
                                                      int M, int N, int Kd) {
    __shared__ bf16_t ldsB[2][128][32] __attribute__((aligned(128)));  // 16 KB

    const int tid  = threadIdx.x;
    const int w    = tid >> 5;
    const int lane = tid & 31;
    const int l    = lane & 15;
    const int h    = lane >> 4;
    const int mw   = w & 3;            // 4 row-groups of waves
    const int nw   = w >> 2;           // 2 col-groups of waves
    const int m0   = blockIdx.x * 128 + mw * 32;
    const int nblk = blockIdx.y * 128;
    const int n0w  = nw * 64;          // wave's N offset within block tile

    // staging assignment: 128 rows x 64B = 512 x 16B chunks; 2 per thread
    const int r0 = tid >> 2,         s0 = tid & 3;
    const int r1 = (tid + 256) >> 2, s1 = (tid + 256) & 3;

    v8f acc[2][4];
#pragma unroll
    for (int rr = 0; rr < 2; ++rr)
#pragma unroll
        for (int f = 0; f < 4; ++f) acc[rr][f] = (v8f){0, 0, 0, 0, 0, 0, 0, 0};

    // prologue: stage B tile for k0 = 0 into buffer 0
    copy16_g2lds(Bt + (size_t)(nblk + r0) * Kd + s0 * 8, &ldsB[0][r0][s0 * 8]);
    copy16_g2lds(Bt + (size_t)(nblk + r1) * Kd + s1 * 8, &ldsB[0][r1][s1 * 8]);

    int buf = 0;
    for (int k0 = 0; k0 < Kd; k0 += 32) {
        if (k0 + 32 < Kd) {  // prefetch next B tile into the other buffer
            const int kn = k0 + 32;
            copy16_g2lds(Bt + (size_t)(nblk + r0) * Kd + kn + s0 * 8, &ldsB[buf ^ 1][r0][s0 * 8]);
            copy16_g2lds(Bt + (size_t)(nblk + r1) * Kd + kn + s1 * 8, &ldsB[buf ^ 1][r1][s1 * 8]);
            wait_async_le2();   // this wave's current-tile pair has landed
        } else {
            wait_async_zero();
        }
        __syncthreads();        // all waves' staging for `buf` complete

        __builtin_prefetch(A + (size_t)(m0 + l) * Kd + k0 + 64, 0, 1);
        v16bf a0 = load_a(A + (size_t)m0 * Kd + k0, Kd, l, h);
        v16bf a1 = load_a(A + (size_t)(m0 + 16) * Kd + k0, Kd, l, h);
#pragma unroll
        for (int f = 0; f < 4; ++f) {
            v16bf b = load_b_nt(&ldsB[buf][n0w + f * 16][0], 32, l, h);
            acc[0][f] = wmma_bf16(a0, b, acc[0][f]);
            acc[1][f] = wmma_bf16(a1, b, acc[1][f]);
        }
        __syncthreads();        // reads done before this buffer is re-targeted
        buf ^= 1;
    }

#pragma unroll
    for (int f = 0; f < 4; ++f) {
        const int n    = nblk + n0w + f * 16 + l;
        const float bv = bias[n];
#pragma unroll
        for (int rr = 0; rr < 2; ++rr) {
#pragma unroll
            for (int r = 0; r < 8; ++r) {
                const int m     = m0 + rr * 16 + r + 8 * h;
                const float val = acc[rr][f][r] + bv;
                if (MODE == 0) {
                    ((float*)outp)[(size_t)m * N + n] = val;
                } else {
                    const int bi = m >> 11;          // m / SS
                    const int s  = m & (SS - 1);
                    const int hh = n >> 6;           // n / HD
                    const int d  = n & (HD - 1);
                    if (MODE == 1)
                        ((bf16_t*)outp)[(((size_t)bi * NH + hh) * SS + s) * HD + d] = (bf16_t)val;
                    else
                        ((bf16_t*)outp)[(((size_t)bi * NH + hh) * HD + d) * SS + s] = (bf16_t)val;
                }
            }
        }
    }
}

// ---------------- causal flash attention ----------------
// grid: (BB*NH, SS/64); block: 128 threads (4 waves, 16 q-rows each)
__global__ __launch_bounds__(128) void attn_kernel(const bf16_t* __restrict__ Q,
                                                   const bf16_t* __restrict__ K,
                                                   const bf16_t* __restrict__ Vt,
                                                   bf16_t* __restrict__ ctx) {
    __shared__ bf16_t ldsP[4][16][32] __attribute__((aligned(64)));

    const int w    = threadIdx.x >> 5;
    const int lane = threadIdx.x & 31;
    const int l    = lane & 15;
    const int h    = lane >> 4;
    const int bh   = blockIdx.x;   // b*NH + head
    const int qt   = blockIdx.y;   // 64-row q tile
    const int q0   = qt * 64 + w * 16;

    const bf16_t* Qp = Q + ((size_t)bh * SS + q0) * HD;
    const bf16_t* Kp = K + (size_t)bh * SS * HD;
    const bf16_t* Vp = Vt + (size_t)bh * HD * SS;

    // Q fragments for the two d-halves (loaded once)
    const v16bf qa0 = load_a(Qp, HD, l, h);
    const v16bf qa1 = load_a(Qp + 32, HD, l, h);

    float mrun[8], lrun[8], alpha[8];
    v8f o[4];
#pragma unroll
    for (int r = 0; r < 8; ++r) { mrun[r] = -3.0e38f; lrun[r] = 0.0f; }
#pragma unroll
    for (int f = 0; f < 4; ++f) o[f] = (v8f){0, 0, 0, 0, 0, 0, 0, 0};

    const float scale = 0.125f;   // 1/sqrt(64)
    const int nkt = 2 * qt + 2;   // key tiles of 32 covering [0, qt*64+64)

    for (int kt = 0; kt < nkt; ++kt) {
        const int kbase = kt * 32;

        // ---- scores S = Q K^T (16 q-rows x 32 keys), reduce over d=64 ----
        v8f s0v = (v8f){0, 0, 0, 0, 0, 0, 0, 0};
        v8f s1v = (v8f){0, 0, 0, 0, 0, 0, 0, 0};
        s0v = wmma_bf16(qa0, load_b_nt(Kp + (size_t)kbase * HD, HD, l, h), s0v);
        s0v = wmma_bf16(qa1, load_b_nt(Kp + (size_t)kbase * HD + 32, HD, l, h), s0v);
        s1v = wmma_bf16(qa0, load_b_nt(Kp + (size_t)(kbase + 16) * HD, HD, l, h), s1v);
        s1v = wmma_bf16(qa1, load_b_nt(Kp + (size_t)(kbase + 16) * HD + 32, HD, l, h), s1v);

        // ---- online softmax (row = r + 8h, col = lane l within half) ----
#pragma unroll
        for (int r = 0; r < 8; ++r) {
            const int qi = q0 + r + 8 * h;
            float v0 = s0v[r] * scale;
            float v1 = s1v[r] * scale;
            if (kbase + l > qi)      v0 = -3.0e38f;   // causal mask
            if (kbase + 16 + l > qi) v1 = -3.0e38f;

            float rm = fmaxf(v0, v1);
#pragma unroll
            for (int off = 8; off >= 1; off >>= 1)
                rm = fmaxf(rm, __shfl_xor(rm, off, 32));   // stays in 16-lane half

            const float mnew = fmaxf(mrun[r], rm);
            alpha[r] = __expf(mrun[r] - mnew);
            mrun[r]  = mnew;

            const float p0 = __expf(v0 - mnew);
            const float p1 = __expf(v1 - mnew);
            float rs = p0 + p1;
#pragma unroll
            for (int off = 8; off >= 1; off >>= 1)
                rs += __shfl_xor(rs, off, 32);
            lrun[r] = lrun[r] * alpha[r] + rs;

            // stage P (C layout) into per-wave LDS as bf16
            ldsP[w][r + 8 * h][l]      = (bf16_t)p0;
            ldsP[w][r + 8 * h][16 + l] = (bf16_t)p1;
        }

        // rescale running context by alpha (rows match C layout)
#pragma unroll
        for (int f = 0; f < 4; ++f)
#pragma unroll
            for (int r = 0; r < 8; ++r) o[f][r] *= alpha[r];

        // re-read P in A-fragment layout (same-wave LDS ops are in-order)
        const v16bf pa = load_a(&ldsP[w][0][0], 32, l, h);

        // ---- ctx += P * V  (V^T stored [d, s]: NT form, k-contiguous) ----
#pragma unroll
        for (int f = 0; f < 4; ++f) {
            v16bf vb = load_b_nt(Vp + (size_t)(f * 16) * SS + kbase, SS, l, h);
            o[f] = wmma_bf16(pa, vb, o[f]);
        }
    }

    // ---- normalize and write ctx back merged as [b, s, h*64+d] bf16 ----
    const int b    = bh >> 4;
    const int head = bh & (NH - 1);
#pragma unroll
    for (int f = 0; f < 4; ++f) {
#pragma unroll
        for (int r = 0; r < 8; ++r) {
            const int q     = q0 + r + 8 * h;
            const float val = o[f][r] / lrun[r];
            ctx[((size_t)b * SS + q) * DM + head * HD + f * 16 + l] = (bf16_t)val;
        }
    }
}

// ---------------- host-side orchestration ----------------
extern "C" void kernel_launch(void* const* d_in, const int* in_sizes, int n_in,
                              void* d_out, int out_size, void* d_ws, size_t ws_size,
                              hipStream_t stream) {
    (void)in_sizes; (void)n_in; (void)out_size; (void)ws_size;

    const float* x  = (const float*)d_in[0];
    const float* wq = (const float*)d_in[1];
    const float* bq = (const float*)d_in[2];
    const float* wk = (const float*)d_in[3];
    const float* bk = (const float*)d_in[4];
    const float* wv = (const float*)d_in[5];
    const float* bv = (const float*)d_in[6];
    const float* wo = (const float*)d_in[7];
    const float* bo = (const float*)d_in[8];

    // workspace layout (bf16), total = 48 MiB
    bf16_t* xb   = (bf16_t*)d_ws;                       // MROWS*DM
    bf16_t* wqb  = xb  + (size_t)MROWS * DM;            // DM*DM
    bf16_t* wkb  = wqb + (size_t)DM * DM;
    bf16_t* wvb  = wkb + (size_t)DM * DM;
    bf16_t* wob  = wvb + (size_t)DM * DM;
    bf16_t* Qb   = wob + (size_t)DM * DM;               // [b,h,s,d]
    bf16_t* Kb   = Qb  + (size_t)MROWS * DM;            // [b,h,s,d]
    bf16_t* Vtb  = Kb  + (size_t)MROWS * DM;            // [b,h,d,s]
    bf16_t* ctxb = Vtb + (size_t)MROWS * DM;            // [b*s, h*d]

    f2bf_kernel<<<512, 256, 0, stream>>>(x,  xb,  MROWS * DM);
    f2bf_kernel<<<256, 256, 0, stream>>>(wq, wqb, DM * DM);
    f2bf_kernel<<<256, 256, 0, stream>>>(wk, wkb, DM * DM);
    f2bf_kernel<<<256, 256, 0, stream>>>(wv, wvb, DM * DM);
    f2bf_kernel<<<256, 256, 0, stream>>>(wo, wob, DM * DM);

    dim3 g(MROWS / 128, DM / 128);
    gemm_nt_kernel<1><<<g, 256, 0, stream>>>(xb, wqb, bq, Qb,  MROWS, DM, DM);
    gemm_nt_kernel<1><<<g, 256, 0, stream>>>(xb, wkb, bk, Kb,  MROWS, DM, DM);
    gemm_nt_kernel<2><<<g, 256, 0, stream>>>(xb, wvb, bv, Vtb, MROWS, DM, DM);

    attn_kernel<<<dim3(BB * NH, SS / 64), 128, 0, stream>>>(Qb, Kb, Vtb, ctxb);

    gemm_nt_kernel<0><<<g, 256, 0, stream>>>(ctxb, wob, bo, d_out, MROWS, DM, DM);
}
